// PyTorchAttentionBaseline_67482526154949
// MI455X (gfx1250) — matrix operand order repaired and checked by reading, not compiled, not run on hardware
//
#include <hip/hip_runtime.h>
#include <hip/hip_bf16.h>

// ---------------------------------------------------------------------------
// MI455X (gfx1250) fused multi-head attention forward.
// bf16 WMMA everywhere (compute-bound: ~206 GFLOP vs ~170MB -> AI ~1000).
// GEMMs: block-cooperative 128x64 tiles; shared B tile double-buffered in LDS
// via async global->LDS (ASYNCcnt). Weights (32MB bf16) fit in the 192MB L2;
// async staging cuts L2 reads 8x.
// ---------------------------------------------------------------------------

typedef __attribute__((ext_vector_type(8)))  __bf16 v8bf;
typedef __attribute__((ext_vector_type(16))) __bf16 v16bf;
typedef __attribute__((ext_vector_type(8)))  float  v8f;
typedef __attribute__((ext_vector_type(8)))  float  f32x8;
typedef __attribute__((ext_vector_type(4)))  int    v4i;

static constexpr int BB = 2;
static constexpr int SS = 2048;
static constexpr int DD = 2048;
static constexpr int HH = 16;
static constexpr int HDI = 128;                        // head dim
static constexpr float SCALE = 0.08838834764831845f;   // 1/sqrt(128)
static constexpr float LOG2E = 1.4426950408889634f;

#if defined(__has_builtin)
#if __has_builtin(__builtin_amdgcn_global_load_async_to_lds_b128) && \
    __has_builtin(__builtin_amdgcn_s_wait_asynccnt)
#define USE_ASYNC_LDS 1
#endif
#endif

#define AS1 __attribute__((address_space(1)))
#define AS3 __attribute__((address_space(3)))

__device__ inline v16bf ld2(const __bf16* p0, const __bf16* p1) {
  const v8bf a = *(const v8bf*)p0;
  const v8bf b = *(const v8bf*)p1;
  return __builtin_shufflevector(a, b, 0,1,2,3,4,5,6,7,8,9,10,11,12,13,14,15);
}

__device__ inline v8f wmma_bf16(v16bf a, v16bf b, v8f c) {
  return __builtin_amdgcn_wmma_f32_16x16x32_bf16(
      /*neg_a=*/false, a, /*neg_b=*/false, b,
      /*c_mod=*/(short)0, c, /*reuse_a=*/false, /*reuse_b=*/false);
}

__device__ inline v8f zero8() {
  v8f z;
#pragma unroll
  for (int i = 0; i < 8; ++i) z[i] = 0.f;
  return z;
}

// 16-byte global -> LDS copy: async (ASYNCcnt) when available, sync fallback.
__device__ inline void cp16(const __bf16* g, __bf16* l) {
#if defined(USE_ASYNC_LDS)
  __builtin_amdgcn_global_load_async_to_lds_b128((AS1 v4i*)g, (AS3 v4i*)l,
                                                 /*offset=*/0, /*cpol=*/0);
#else
  *(v8bf*)l = *(const v8bf*)g;
#endif
}

__device__ inline void wait_async_copies() {
#if defined(USE_ASYNC_LDS)
  __builtin_amdgcn_s_wait_asynccnt(0);
#endif
}

// ---------------------------------------------------------------------------
// f32 -> bf16 conversion (weights only; activations converted in-register)
// ---------------------------------------------------------------------------
__global__ __launch_bounds__(256) void cvt_f32_bf16(const float* __restrict__ in,
                                                    __bf16* __restrict__ out, int n) {
  int i = (blockIdx.x * blockDim.x + threadIdx.x) * 4;
  if (i + 3 < n) {
    const float4 v = *(const float4*)(in + i);
    out[i + 0] = (__bf16)v.x;
    out[i + 1] = (__bf16)v.y;
    out[i + 2] = (__bf16)v.z;
    out[i + 3] = (__bf16)v.w;
  }
}

// ---------------------------------------------------------------------------
// GEMM: Y[M,N] = X[M,K] @ W[N,K]^T + bias[N]
// Block = 8 waves = 128x64 output tile (wave w owns rows bm*128+w*16..+15).
// Shared B tile (64 cols x 32 K, bf16, LDS stride 40) double-buffered via
// async global->LDS; each of the 256 threads copies one 16B chunk per K step.
// ---------------------------------------------------------------------------
template <int AF32, int F32OUT, int TRANSOUT>
__global__ __launch_bounds__(256) void gemm_wmma(const void* __restrict__ Ain,
                                                 const __bf16* __restrict__ W,
                                                 const float* __restrict__ bias,
                                                 void* __restrict__ Y,
                                                 int M, int N, int K, int Sb) {
  constexpr int BSTR = 40;                 // LDS col stride (bf16) for bank spread
  __shared__ __bf16 bsh[2][64 * BSTR];     // 2 x 5KB double buffer

  const int t    = threadIdx.x;
  const int lane = t & 31;
  const int l16  = lane & 15;
  const int lhi  = (lane >> 4) & 1;
  const int ws   = t >> 5;
  const int NT   = N >> 6;
  const int bm   = blockIdx.x / NT;
  const int nt   = blockIdx.x % NT;
  const int arow = bm * 128 + ws * 16 + l16;

  const int ccol = t >> 2;                 // column this thread copies (0..63)
  const int cchk = (t & 3) * 8;            // K-element offset of its 16B chunk
  const __bf16* wsrc = W + (size_t)(nt * 64 + ccol) * K + cchk;

  v8f acc[4];
#pragma unroll
  for (int s = 0; s < 4; ++s) acc[s] = zero8();

  const int nchunks = K >> 5;
  cp16(wsrc, &bsh[0][ccol * BSTR + cchk]);               // prologue: chunk 0

  for (int i = 0; i < nchunks; ++i) {
    const int kk = i << 5;
    wait_async_copies();                                 // my copy landed
    __syncthreads();                                     // everyone's copy landed
    if (i + 1 < nchunks)                                 // overlap next copy
      cp16(wsrc + kk + 32, &bsh[(i + 1) & 1][ccol * BSTR + cchk]);

    v16bf a;
    if constexpr (AF32) {
      const float* ap = (const float*)Ain + (size_t)arow * K + kk;
      const f32x8 f0 = *(const f32x8*)(ap + (lhi ? 8 : 0));
      const f32x8 f1 = *(const f32x8*)(ap + 16 + (lhi ? 8 : 0));
#pragma unroll
      for (int j = 0; j < 8; ++j) { a[j] = (__bf16)f0[j]; a[8 + j] = (__bf16)f1[j]; }
      __builtin_prefetch((const void*)(ap + 64), 0, 1);
    } else {
      const __bf16* ap = (const __bf16*)Ain + (size_t)arow * K + kk;
      a = ld2(ap + (lhi ? 8 : 0), ap + 16 + (lhi ? 8 : 0));
      __builtin_prefetch((const void*)(ap + 64), 0, 1);
    }

    const __bf16* bbase = &bsh[i & 1][0];
#pragma unroll
    for (int sub = 0; sub < 4; ++sub) {
      const __bf16* bp = bbase + (sub * 16 + l16) * BSTR + lhi * 8;
      acc[sub] = wmma_bf16(a, ld2(bp, bp + 16), acc[sub]);
    }
    __syncthreads();                                     // done reading buf i&1
  }

#pragma unroll
  for (int sub = 0; sub < 4; ++sub) {
    const int col = nt * 64 + sub * 16 + l16;
    const float bv = bias[col];
#pragma unroll
    for (int r = 0; r < 8; ++r) {
      const int row = bm * 128 + ws * 16 + r + lhi * 8;  // C layout row mapping
      const float v = acc[sub][r] + bv;
      if constexpr (F32OUT) {
        ((float*)Y)[(size_t)row * N + col] = v;
      } else if constexpr (TRANSOUT) {
        const int bidx = row / Sb, s = row % Sb;
        ((__bf16*)Y)[((size_t)bidx * N + col) * Sb + s] = (__bf16)v;
      } else {
        ((__bf16*)Y)[(size_t)row * N + col] = (__bf16)v;
      }
    }
  }
}

// ---------------------------------------------------------------------------
// Flash attention: one wave per 16-row Q tile of one (b,h). Online softmax.
// Q/K stored [B,S,D] bf16; V stored transposed [B,D,S] bf16.
// P staged through per-wave LDS (bf16, stride 40) to convert C-layout into
// the A-operand layout via two ds_load_b128 per lane.
// ---------------------------------------------------------------------------
__global__ __launch_bounds__(256) void flash_attn_kernel(const __bf16* __restrict__ Qp,
                                                         const __bf16* __restrict__ Kp,
                                                         const __bf16* __restrict__ Vt,
                                                         __bf16* __restrict__ AO) {
  constexpr int NQT = SS / 16;
  __shared__ __bf16 plds[8][16 * 40];

  const int lane = threadIdx.x & 31;
  const int l16  = lane & 15;
  const int lhi  = (lane >> 4) & 1;
  const int ws   = threadIdx.x >> 5;
  const int w    = blockIdx.x * 8 + ws;
  const int b    = w / (HH * NQT);
  const int h    = (w / NQT) % HH;
  const int qt   = w % NQT;
  const int qbase = qt * 16;

  v16bf qa[4];
  {
    const __bf16* qrow = Qp + ((size_t)(b * SS + qbase + l16)) * DD + h * HDI;
#pragma unroll
    for (int c = 0; c < 4; ++c) {
      const __bf16* p = qrow + c * 32 + (lhi ? 8 : 0);
      qa[c] = ld2(p, p + 16);
    }
  }

  v8f o[8];
#pragma unroll
  for (int n = 0; n < 8; ++n) o[n] = zero8();
  float rm[8], rs[8];
#pragma unroll
  for (int r = 0; r < 8; ++r) { rm[r] = -1e30f; rs[r] = 0.f; }

  const float sft = SCALE * LOG2E;
  const int nkb = (qbase + 16 + 31) >> 5;

  for (int kb = 0; kb < nkb; ++kb) {
    const int kbase = kb * 32;
    v8f sc0 = zero8(), sc1 = zero8();
#pragma unroll
    for (int c = 0; c < 4; ++c) {
      const int koff = h * HDI + c * 32 + (lhi ? 8 : 0);
      const __bf16* k0 = Kp + (size_t)(b * SS + kbase + l16) * DD + koff;
      const __bf16* k1 = Kp + (size_t)(b * SS + kbase + 16 + l16) * DD + koff;
      sc0 = wmma_bf16(qa[c], ld2(k0, k0 + 16), sc0);
      sc1 = wmma_bf16(qa[c], ld2(k1, k1 + 16), sc1);
    }

#pragma unroll
    for (int r = 0; r < 8; ++r) {
      const int grow = qbase + r + lhi * 8;
      float s0 = sc0[r] * sft;
      float s1 = sc1[r] * sft;
      if (kbase + l16 > grow)      s0 = -1e30f;
      if (kbase + 16 + l16 > grow) s1 = -1e30f;
      float bm = fmaxf(s0, s1);
#pragma unroll
      for (int m = 1; m < 16; m <<= 1) bm = fmaxf(bm, __shfl_xor(bm, m, 32));
      const float nm   = fmaxf(rm[r], bm);
      const float corr = exp2f(rm[r] - nm);
      rm[r] = nm;
      const float p0 = exp2f(s0 - nm);
      const float p1 = exp2f(s1 - nm);
      float bs = p0 + p1;
#pragma unroll
      for (int m = 1; m < 16; m <<= 1) bs += __shfl_xor(bs, m, 32);
      rs[r] = rs[r] * corr + bs;
#pragma unroll
      for (int n = 0; n < 8; ++n) o[n][r] *= corr;
      plds[ws][(r + lhi * 8) * 40 + l16]      = (__bf16)p0;
      plds[ws][(r + lhi * 8) * 40 + 16 + l16] = (__bf16)p1;
    }
    asm volatile("s_wait_dscnt 0" ::: "memory");   // same-wave LDS RAW fence

    const v16bf pa = ld2(&plds[ws][l16 * 40 + (lhi ? 8 : 0)],
                         &plds[ws][l16 * 40 + 16 + (lhi ? 8 : 0)]);

#pragma unroll
    for (int n = 0; n < 8; ++n) {
      const __bf16* vp = Vt + ((size_t)(b * DD + h * HDI + n * 16 + l16)) * SS
                            + kbase + (lhi ? 8 : 0);
      o[n] = wmma_bf16(pa, ld2(vp, vp + 16), o[n]);
    }
  }

#pragma unroll
  for (int r = 0; r < 8; ++r) {
    const float inv = 1.0f / rs[r];
    const size_t rowoff = ((size_t)(b * SS + qbase + r + lhi * 8)) * DD + h * HDI;
#pragma unroll
    for (int n = 0; n < 8; ++n)
      AO[rowoff + n * 16 + l16] = (__bf16)(o[n][r] * inv);
  }
}

// ---------------------------------------------------------------------------
extern "C" void kernel_launch(void* const* d_in, const int* in_sizes, int n_in,
                              void* d_out, int out_size, void* d_ws, size_t ws_size,
                              hipStream_t stream) {
  const float* query = (const float*)d_in[0];
  const float* keyi  = (const float*)d_in[1];
  const float* value = (const float*)d_in[2];
  const float* Wq = (const float*)d_in[3];  const float* bq = (const float*)d_in[4];
  const float* Wk = (const float*)d_in[5];  const float* bk = (const float*)d_in[6];
  const float* Wv = (const float*)d_in[7];  const float* bv = (const float*)d_in[8];
  const float* Wo = (const float*)d_in[9];  const float* bo = (const float*)d_in[10];

  const size_t szw = (size_t)DD * DD;
  const size_t szx = (size_t)BB * SS * DD;

  char* p = (char*)d_ws;
  __bf16* wqb = (__bf16*)p; p += szw * 2;
  __bf16* wkb = (__bf16*)p; p += szw * 2;
  __bf16* wvb = (__bf16*)p; p += szw * 2;
  __bf16* wob = (__bf16*)p; p += szw * 2;
  __bf16* Qp  = (__bf16*)p; p += szx * 2;
  __bf16* Kp  = (__bf16*)p; p += szx * 2;
  __bf16* Vt  = (__bf16*)p; p += szx * 2;   // [B,D,S]
  __bf16* AO  = (__bf16*)p; p += szx * 2;

  cvt_f32_bf16<<<(int)(szw / 1024), 256, 0, stream>>>(Wq, wqb, (int)szw);
  cvt_f32_bf16<<<(int)(szw / 1024), 256, 0, stream>>>(Wk, wkb, (int)szw);
  cvt_f32_bf16<<<(int)(szw / 1024), 256, 0, stream>>>(Wv, wvb, (int)szw);
  cvt_f32_bf16<<<(int)(szw / 1024), 256, 0, stream>>>(Wo, wob, (int)szw);

  const int M = BB * SS, N = DD, K = DD;
  const int gemm_blocks = (M / 128) * (N / 64);        // 32*32 = 1024 blocks

  gemm_wmma<1, 0, 0><<<gemm_blocks, 256, 0, stream>>>(query, wqb, bq, Qp, M, N, K, SS);
  gemm_wmma<1, 0, 0><<<gemm_blocks, 256, 0, stream>>>(keyi,  wkb, bk, Kp, M, N, K, SS);
  gemm_wmma<1, 0, 1><<<gemm_blocks, 256, 0, stream>>>(value, wvb, bv, Vt, M, N, K, SS);

  const int fa_blocks = BB * HH * (SS / 16) / 8;       // 512 blocks
  flash_attn_kernel<<<fa_blocks, 256, 0, stream>>>(Qp, Kp, Vt, AO);

  gemm_wmma<0, 1, 0><<<gemm_blocks, 256, 0, stream>>>(AO, wob, bo, d_out, M, N, K, SS);
}
